// MultiLayerNormGRU_77481210020041
// MI455X (gfx1250) — compile-verified, matrix operand order checked by reading, not run on hardware
//
#include <hip/hip_runtime.h>

// ---------------------------------------------------------------------------
// MultiLayerNormGRU on gfx1250 (MI455X, wave32, WMMA)
// B=512, T=512, D_IN=64, H=256, OUT=64
// One workgroup = 16 batch rows for all T steps; 512 threads = 16 waves.
// All GEMMs via v_wmma_f32_16x16x32_bf16; weights pre-packed to bf16 lane
// fragments in d_ws so B operands are two global_load_b128 per lane.
// Weight-fragment base pointers are (a) typed as address_space(1) so they
// lower to global_load (not flat_load), and (b) laundered through an empty
// asm each timestep so LLVM cannot hoist the loop-invariant weight loads out
// of the 512-iteration loop and spill them to scratch.
// ---------------------------------------------------------------------------

#define TSTEPS 512
#define BATCH  512
#define DIN    64
#define HDIM   256
#define OUTD   64

typedef __attribute__((ext_vector_type(8)))  float  v8f;
typedef __attribute__((ext_vector_type(8)))  __bf16 v8bf;
typedef __attribute__((ext_vector_type(16))) __bf16 v16bf;

// global (addrspace 1) flavors for operand streaming
typedef __attribute__((address_space(1))) const unsigned short gu16;
typedef __attribute__((address_space(1))) const v8bf          gv8bf;

__device__ __forceinline__ const gu16* to_global(const unsigned short* p)
{
    // global aperture is identity-mapped inside flat space: raw bitcast is exact
    return (const gu16*)(unsigned long long)p;
}

// LDS layout (bytes); strides padded to dodge bank conflicts
#define A0S 328                  // bf16 stride for [16 x 320] concat(x, h0) buffer
#define A1S 520                  // bf16 stride for [16 x 512] concat(h0, h1) buffer
#define GS  516                  // f32  stride for [16 x 512] pre-activation buffer

#define OFF_A0 0                 // 16*328*2 = 10496
#define OFF_A1 10496             // 16*520*2 = 16640
#define OFF_G  27136             // 16*516*4 = 33024
#define OFF_Z  60160             // 16*256*4 = 16384
#define OFF_H0 76544             // 16*256*4 = 16384
#define OFF_H1 92928             // 16*256*4 = 16384
#define OFF_P  109312            // 4608 floats of bias/LN params = 18432
#define SMEM_BYTES 127744

// param block offsets (floats) inside P
#define P_BG0  0
#define P_GNW0 512
#define P_GNB0 1024
#define P_BC0  1536
#define P_CNW0 1792
#define P_CNB0 2048
#define P_BG1  2304
#define P_GNW1 2816
#define P_GNB1 3328
#define P_BC1  3840
#define P_CNW1 4096
#define P_CNB1 4352

// ---------------------------------------------------------------------------
// One 16x16 output tile, K = KT*32, A from LDS, B from packed global frags.
// Fragment layout per 32x16 (KxN) tile: lane l holds column n = l%16, K
// values {kt*32 + (l/16)*8 + 0..7} and {kt*32 + 16 + (l/16)*8 + 0..7}
// -> two 16B loads per lane for both A and B.
// ---------------------------------------------------------------------------
template <int KT>
__device__ __forceinline__ v8f mm_tile(const __bf16* __restrict__ Arow,
                                       const gu16* __restrict__ Fp)
{
    v8f acc = {0.f, 0.f, 0.f, 0.f, 0.f, 0.f, 0.f, 0.f};
#pragma unroll
    for (int kt = 0; kt < KT; ++kt) {
        v8bf alo = *(const v8bf*)(Arow + kt * 32);
        v8bf ahi = *(const v8bf*)(Arow + kt * 32 + 16);
        v8bf blo = *(const gv8bf*)(Fp + kt * 512);
        v8bf bhi = *(const gv8bf*)(Fp + kt * 512 + 8);
        v16bf a = __builtin_shufflevector(alo, ahi, 0,1,2,3,4,5,6,7,8,9,10,11,12,13,14,15);
        v16bf b = __builtin_shufflevector(blo, bhi, 0,1,2,3,4,5,6,7,8,9,10,11,12,13,14,15);
        acc = __builtin_amdgcn_wmma_f32_16x16x32_bf16(false, a, false, b,
                                                      (short)0, acc, false, false);
    }
    return acc;
}

// D tile (lane l, vgpr r) = element (row r + 8*(l/16), col l%16) -> LDS f32
__device__ __forceinline__ void store_d(float* __restrict__ G, int nt, int lane, v8f acc)
{
    const int mrow = lane & 15, hf = lane >> 4;
    float* p = G + (size_t)(8 * hf) * GS + nt * 16 + mrow;
#pragma unroll
    for (int r = 0; r < 8; ++r) p[(size_t)r * GS] = acc[r];
}

__device__ __forceinline__ float fast_sigmoid(float u)
{
    return 1.f / (1.f + __expf(-u));
}

__device__ __forceinline__ float fast_tanh(float u)
{
    // tanh(u) = 1 - 2/(exp(2u)+1)
    return fmaf(-2.f, 1.f / (__expf(2.f * u) + 1.f), 1.f);
}

// LayerNorm over 512 gate columns (one wave per row), then sigmoid.
// cols 0..255 (r-gate): write r*h_prev as bf16 into rdest.
// cols 256..511 (z-gate): stash z in zrow.
__device__ __forceinline__ void ln_gate(const float* __restrict__ Grow,
                                        const float* __restrict__ bias,
                                        const float* __restrict__ nw,
                                        const float* __restrict__ nb,
                                        const float* __restrict__ hrow,
                                        __bf16* __restrict__ rdest,
                                        float* __restrict__ zrow, int lane)
{
    const int c0 = lane * 16;
    float vals[16];
    float s = 0.f, s2 = 0.f;
#pragma unroll
    for (int i = 0; i < 16; ++i) {
        float v = Grow[c0 + i] + bias[c0 + i];
        vals[i] = v; s += v; s2 += v * v;
    }
#pragma unroll
    for (int off = 16; off > 0; off >>= 1) {
        s  += __shfl_xor(s,  off, 32);
        s2 += __shfl_xor(s2, off, 32);
    }
    const float mean = s * (1.f / 512.f);
    const float var  = s2 * (1.f / 512.f) - mean * mean;
    const float rstd = rsqrtf(var + 1e-5f);
#pragma unroll
    for (int i = 0; i < 16; ++i) {
        const int c = c0 + i;
        const float u  = (vals[i] - mean) * rstd * nw[c] + nb[c];
        const float sg = fast_sigmoid(u);
        if (c0 < HDIM) rdest[c] = (__bf16)(sg * hrow[c]);     // r * h_prev
        else           zrow[c - HDIM] = sg;                   // z
    }
}

// LayerNorm over 256 candidate columns + tanh + GRU state update.
__device__ __forceinline__ void ln_cand_update(const float* __restrict__ Grow,
                                               const float* __restrict__ bias,
                                               const float* __restrict__ nw,
                                               const float* __restrict__ nb,
                                               const float* __restrict__ zrow,
                                               float* __restrict__ hrow,
                                               __bf16* __restrict__ d1,
                                               __bf16* __restrict__ d2, int lane)
{
    const int c0 = lane * 8;
    float vals[8];
    float s = 0.f, s2 = 0.f;
#pragma unroll
    for (int i = 0; i < 8; ++i) {
        float v = Grow[c0 + i] + bias[c0 + i];
        vals[i] = v; s += v; s2 += v * v;
    }
#pragma unroll
    for (int off = 16; off > 0; off >>= 1) {
        s  += __shfl_xor(s,  off, 32);
        s2 += __shfl_xor(s2, off, 32);
    }
    const float mean = s * (1.f / 256.f);
    const float var  = s2 * (1.f / 256.f) - mean * mean;
    const float rstd = rsqrtf(var + 1e-5f);
#pragma unroll
    for (int i = 0; i < 8; ++i) {
        const int c = c0 + i;
        const float u = (vals[i] - mean) * rstd * nw[c] + nb[c];
        const float n = fast_tanh(u);
        const float z = zrow[c];
        const float hp = hrow[c];
        const float hn = fmaf(z, n - hp, hp);     // (1-z)*h + z*n
        hrow[c] = hn;
        const __bf16 hb = (__bf16)hn;
        d1[c] = hb;
        if (d2) d2[c] = hb;
    }
}

// ---------------------------------------------------------------------------
// Weight pre-pack: fp32 row-major [N,K] -> bf16 fragments, frag f = nt*KT+kt,
// 32 lanes x 16 halfs per frag (512 halfs = 1 KB).
// ---------------------------------------------------------------------------
__global__ void pack_w_kernel(const float* __restrict__ W,
                              unsigned short* __restrict__ outw,
                              int K, int ktiles)
{
    const int f    = blockIdx.x;
    const int lane = threadIdx.x;          // 32
    const int nt = f / ktiles, kt = f % ktiles;
    const int n  = nt * 16 + (lane & 15);
    const int hk = (lane >> 4) * 8;
    const float* wr = W + (size_t)n * K + kt * 32 + hk;
    __bf16* o = (__bf16*)(outw + (size_t)f * 512 + lane * 16);
#pragma unroll
    for (int j = 0; j < 16; ++j) {
        const int k = ((j >> 3) * 16) + (j & 7);
        o[j] = (__bf16)wr[k];
    }
}

// ---------------------------------------------------------------------------
// Persistent GRU kernel: 32 blocks x 512 threads (16 waves, wave32).
// ---------------------------------------------------------------------------
__global__ __launch_bounds__(512) void gru_persist(
    const float* __restrict__ x,
    const unsigned short* __restrict__ fg0_in, const unsigned short* __restrict__ fc0_in,
    const unsigned short* __restrict__ fg1_in, const unsigned short* __restrict__ fc1_in,
    const unsigned short* __restrict__ ffc_in,
    const float* __restrict__ bg0, const float* __restrict__ gnw0, const float* __restrict__ gnb0,
    const float* __restrict__ bc0, const float* __restrict__ cnw0, const float* __restrict__ cnb0,
    const float* __restrict__ bg1, const float* __restrict__ gnw1, const float* __restrict__ gnb1,
    const float* __restrict__ bc1, const float* __restrict__ cnw1, const float* __restrict__ cnb1,
    const float* __restrict__ fcb,
    float* __restrict__ out)
{
    extern __shared__ char smem[];
    __bf16* A0 = (__bf16*)(smem + OFF_A0);   // [16][A0S] concat(x_t, h0|r*h0) bf16
    __bf16* A1 = (__bf16*)(smem + OFF_A1);   // [16][A1S] concat(h0, h1|r*h1) bf16
    float*  G  = (float*) (smem + OFF_G);    // [16][GS]  GEMM pre-activations
    float*  Z  = (float*) (smem + OFF_Z);    // [16][256] z gate
    float*  H0 = (float*) (smem + OFF_H0);   // [16][256] fp32 hidden state L0
    float*  H1 = (float*) (smem + OFF_H1);   // [16][256] fp32 hidden state L1
    float*  P  = (float*) (smem + OFF_P);    // bias / LN params

    const int tid  = threadIdx.x;
    const int lane = tid & 31;
    const int wv   = tid >> 5;               // 0..15
    const int mrow = lane & 15;
    const int hf   = lane >> 4;
    const int b0   = blockIdx.x * 16;

    // stage small params to LDS once
    { int i = tid;
      if (i < 512) { P[P_BG0+i]=bg0[i]; P[P_GNW0+i]=gnw0[i]; P[P_GNB0+i]=gnb0[i];
                     P[P_BG1+i]=bg1[i]; P[P_GNW1+i]=gnw1[i]; P[P_GNB1+i]=gnb1[i]; }
      if (i < 256) { P[P_BC0+i]=bc0[i]; P[P_CNW0+i]=cnw0[i]; P[P_CNB0+i]=cnb0[i];
                     P[P_BC1+i]=bc1[i]; P[P_CNW1+i]=cnw1[i]; P[P_CNB1+i]=cnb1[i]; } }
    // zero state + concat buffers (h(0) = 0)
    for (int i = tid; i < 16 * A0S; i += 512) A0[i] = (__bf16)0.f;
    for (int i = tid; i < 16 * A1S; i += 512) A1[i] = (__bf16)0.f;
    for (int i = tid; i < 16 * HDIM; i += 512) { H0[i] = 0.f; H1[i] = 0.f; }
    __syncthreads();

    const __bf16* Arow0 = A0 + mrow * A0S + hf * 8;
    const __bf16* Arow1 = A1 + mrow * A1S + hf * 8;

    // addrspace(1) weight-fragment bases (laundered per step below)
    const gu16* fg0 = to_global(fg0_in);
    const gu16* fc0 = to_global(fc0_in);
    const gu16* fg1 = to_global(fg1_in);
    const gu16* fc1 = to_global(fc1_in);

    for (int t = 0; t < TSTEPS; ++t) {
        // Opaque redefinition of the weight-fragment bases: blocks LICM from
        // hoisting 512 iterations' worth of loop-invariant weight loads into
        // registers (which would spill to scratch). Costs one no-op per step.
        asm volatile("" : "+s"(fg0), "+s"(fc0), "+s"(fg1), "+s"(fc1));

        // stage x_t -> bf16 into A0[:, 0:64] (one packed b32 store per thread)
        {
            const int m = tid >> 5;
            const int c = (tid & 31) * 2;
            const float2 v = *(const float2*)(x + ((size_t)(b0 + m) * TSTEPS + t) * DIN + c);
            union { __bf16 b[2]; unsigned int u; } pk;
            pk.b[0] = (__bf16)v.x;
            pk.b[1] = (__bf16)v.y;
            *(unsigned int*)(A0 + m * A0S + c) = pk.u;
        }
        __syncthreads();

        // ---- layer0 gates: [16x320] @ [320x512] ----
#pragma unroll
        for (int nn = 0; nn < 2; ++nn) {
            const int nt = wv * 2 + nn;
            v8f acc = mm_tile<10>(Arow0, fg0 + ((size_t)nt * 10) * 512 + lane * 16);
            store_d(G, nt, lane, acc);
        }
        __syncthreads();
        ln_gate(G + wv * GS, P + P_BG0, P + P_GNW0, P + P_GNB0,
                H0 + wv * HDIM, A0 + wv * A0S + DIN, Z + wv * HDIM, lane);
        __syncthreads();

        // ---- layer0 candidate: [16x320] @ [320x256] ----
        {
            v8f acc = mm_tile<10>(Arow0, fc0 + ((size_t)wv * 10) * 512 + lane * 16);
            store_d(G, wv, lane, acc);
        }
        __syncthreads();
        ln_cand_update(G + wv * GS, P + P_BC0, P + P_CNW0, P + P_CNB0,
                       Z + wv * HDIM, H0 + wv * HDIM,
                       A1 + wv * A1S,            // h0 -> layer1 input (cols 0..255)
                       A0 + wv * A0S + DIN,      // h0 -> next-step layer0 input
                       lane);
        __syncthreads();

        // ---- layer1 gates: [16x512] @ [512x512] ----
#pragma unroll
        for (int nn = 0; nn < 2; ++nn) {
            const int nt = wv * 2 + nn;
            v8f acc = mm_tile<16>(Arow1, fg1 + ((size_t)nt * 16) * 512 + lane * 16);
            store_d(G, nt, lane, acc);
        }
        __syncthreads();
        ln_gate(G + wv * GS, P + P_BG1, P + P_GNW1, P + P_GNB1,
                H1 + wv * HDIM, A1 + wv * A1S + HDIM, Z + wv * HDIM, lane);
        __syncthreads();

        // ---- layer1 candidate: [16x512] @ [512x256] ----
        {
            v8f acc = mm_tile<16>(Arow1, fc1 + ((size_t)wv * 16) * 512 + lane * 16);
            store_d(G, wv, lane, acc);
        }
        __syncthreads();
        ln_cand_update(G + wv * GS, P + P_BC1, P + P_CNW1, P + P_CNB1,
                       Z + wv * HDIM, H1 + wv * HDIM,
                       A1 + wv * A1S + HDIM, nullptr, lane);
        __syncthreads();
    }

    // ---- output head: out = h1 @ fcW^T + fcb  ([16x256] @ [256x64]) ----
    if (wv < 4) {
        const int nt = wv;
        const gu16* ffc = to_global(ffc_in);
        v8f acc = mm_tile<8>(Arow1 + HDIM, ffc + ((size_t)nt * 8) * 512 + lane * 16);
        const int n = nt * 16 + mrow;
        const float bb = fcb[n];
#pragma unroll
        for (int r = 0; r < 8; ++r)
            out[(size_t)(b0 + r + 8 * hf) * OUTD + n] = acc[r] + bb;
    }
}

// ---------------------------------------------------------------------------
extern "C" void kernel_launch(void* const* d_in, const int* in_sizes, int n_in,
                              void* d_out, int out_size, void* d_ws, size_t ws_size,
                              hipStream_t stream)
{
    (void)in_sizes; (void)n_in; (void)out_size; (void)ws_size;

    const float* x    = (const float*)d_in[0];
    const float* Wg0  = (const float*)d_in[1];
    const float* bg0  = (const float*)d_in[2];
    const float* gnw0 = (const float*)d_in[3];
    const float* gnb0 = (const float*)d_in[4];
    const float* Wc0  = (const float*)d_in[5];
    const float* bc0  = (const float*)d_in[6];
    const float* cnw0 = (const float*)d_in[7];
    const float* cnb0 = (const float*)d_in[8];
    const float* Wg1  = (const float*)d_in[9];
    const float* bg1  = (const float*)d_in[10];
    const float* gnw1 = (const float*)d_in[11];
    const float* gnb1 = (const float*)d_in[12];
    const float* Wc1  = (const float*)d_in[13];
    const float* bc1  = (const float*)d_in[14];
    const float* cnw1 = (const float*)d_in[15];
    const float* cnb1 = (const float*)d_in[16];
    const float* fcW  = (const float*)d_in[17];
    const float* fcb  = (const float*)d_in[18];
    float* out = (float*)d_out;

    // packed bf16 weight fragments in workspace (~1.25 MB total)
    unsigned short* ws  = (unsigned short*)d_ws;
    unsigned short* fg0 = ws;                              // 320 frags (N=512,K=320)
    unsigned short* fc0 = fg0 + (size_t)320 * 512;         // 160 frags (N=256,K=320)
    unsigned short* fg1 = fc0 + (size_t)160 * 512;         // 512 frags (N=512,K=512)
    unsigned short* fc1 = fg1 + (size_t)512 * 512;         // 256 frags (N=256,K=512)
    unsigned short* ffc = fc1 + (size_t)256 * 512;         //  32 frags (N=64, K=256)

    pack_w_kernel<<<320, 32, 0, stream>>>(Wg0, fg0, 320, 10);
    pack_w_kernel<<<160, 32, 0, stream>>>(Wc0, fc0, 320, 10);
    pack_w_kernel<<<512, 32, 0, stream>>>(Wg1, fg1, 512, 16);
    pack_w_kernel<<<256, 32, 0, stream>>>(Wc1, fc1, 512, 16);
    pack_w_kernel<<< 32, 32, 0, stream>>>(fcW, ffc, 256, 8);

    static_assert(SMEM_BYTES <= 160 * 1024, "LDS budget");
    hipFuncSetAttribute((const void*)gru_persist,
                        hipFuncAttributeMaxDynamicSharedMemorySize, SMEM_BYTES);
    gru_persist<<<BATCH / 16, 512, SMEM_BYTES, stream>>>(
        x, fg0, fc0, fg1, fc1, ffc,
        bg0, gnw0, gnb0, bc0, cnw0, cnb0,
        bg1, gnw1, gnb1, bc1, cnw1, cnb1,
        fcb, out);
}
